// CausalSelfAttention_4131758538978
// MI455X (gfx1250) — matrix operand order repaired and checked by reading, not compiled, not run on hardware
//
#include <hip/hip_runtime.h>
#include <hip/hip_bf16.h>
#include <stdint.h>

// ---------------------------------------------------------------------------
// CDNA5 (gfx1250) causal self-attention, bf16 WMMA pipeline.
//   B=2, T=2048, C=1024, H=16, D=64
// ---------------------------------------------------------------------------

typedef __bf16 bf16_t;
typedef __attribute__((ext_vector_type(16))) __bf16 v16bf;
typedef __attribute__((ext_vector_type(8)))  __bf16 v8bf;
typedef __attribute__((ext_vector_type(4)))  __bf16 v4bf;
typedef __attribute__((ext_vector_type(8)))  float  v8f;

#define B_    2
#define T_    2048
#define C_    1024
#define NH_   16
#define HD_   64
#define BH_   (B_ * NH_)     // 32
#define MTOK  (B_ * T_)      // 4096

// D = A(16x32 bf16) * B(32x16 bf16) + C(16x16 f32)
__device__ __forceinline__ v8f wmma_bf16(v16bf a, v16bf b, v8f c) {
  return __builtin_amdgcn_wmma_f32_16x16x32_bf16(
      /*neg_a=*/false, a, /*neg_b=*/false, b,
      /*c_mod=*/(short)0, c, /*reuse_a=*/false, /*reuse_b=*/false);
}

// A-fragment: 16(M) x 32(K) tile of a row-major bf16 matrix.
// ISA layout: lane = M row (lanes 16-31 repeat rows with upper K half);
// per lane two contiguous 16-byte chunks: cols [g*8, g*8+8) and [16+g*8, ...).
__device__ __forceinline__ v16bf load_a_frag(const bf16_t* __restrict__ src,
                                             int ld, int lane) {
  const int m = lane & 15, g = lane >> 4;
  const bf16_t* p = src + m * ld + g * 8;
  v8bf lo = *(const v8bf*)(p);
  v8bf hi = *(const v8bf*)(p + 16);
  v16bf f;
#pragma unroll
  for (int i = 0; i < 8; ++i) { f[i] = lo[i]; f[i + 8] = hi[i]; }
  return f;
}

// B-fragment: 32(K) x 16(N), built from a row-major [N, K] matrix (i.e. the
// operand we multiply as its transpose). Lane = N column; two contiguous
// 16-byte chunks: rows src[n], cols [g*8, g*8+8) and [16+g*8, ...).
__device__ __forceinline__ v16bf load_b_frag(const bf16_t* __restrict__ src,
                                             int ld, int lane) {
  const int n = lane & 15, g = lane >> 4;
  const bf16_t* p = src + n * ld + g * 8;
  v8bf lo = *(const v8bf*)(p);
  v8bf hi = *(const v8bf*)(p + 16);
  v16bf f;
#pragma unroll
  for (int i = 0; i < 8; ++i) { f[i] = lo[i]; f[i + 8] = hi[i]; }
  return f;
}

// 16-lane (matrix-row) reductions matching the C/D layout (row spans 16 lanes)
__device__ __forceinline__ float rowmax16(float v) {
  v = fmaxf(v, __shfl_xor(v, 8, 16));
  v = fmaxf(v, __shfl_xor(v, 4, 16));
  v = fmaxf(v, __shfl_xor(v, 2, 16));
  v = fmaxf(v, __shfl_xor(v, 1, 16));
  return v;
}
__device__ __forceinline__ float rowsum16(float v) {
  v += __shfl_xor(v, 8, 16);
  v += __shfl_xor(v, 4, 16);
  v += __shfl_xor(v, 2, 16);
  v += __shfl_xor(v, 1, 16);
  return v;
}

// ---------------------------------------------------------------------------
// Kernel 0: fp32 -> bf16 conversion (x, W_qkv, W_proj)
// ---------------------------------------------------------------------------
__global__ __launch_bounds__(256) void cvt_f32_bf16_kernel(
    const float* __restrict__ in, bf16_t* __restrict__ out, int n) {
  int i = (blockIdx.x * 256 + threadIdx.x) * 4;
  if (i + 3 < n) {
    float4 v = *(const float4*)(in + i);
    v4bf o;
    o[0] = (bf16_t)v.x; o[1] = (bf16_t)v.y;
    o[2] = (bf16_t)v.z; o[3] = (bf16_t)v.w;
    *(v4bf*)(out + i) = o;
  }
}

// ---------------------------------------------------------------------------
// Kernel 1: QKV projection.  [4096,1024] x [3072,1024]^T + bias.
// Writes Q,K as bf16 [BH, T, 64] and V transposed as bf16 [BH, 64, T].
// Block: 256 thr (8 waves) -> 128(M) x 128(N) tile; wave = 32x64.
// ---------------------------------------------------------------------------
__global__ __launch_bounds__(256) void qkv_gemm_kernel(
    const bf16_t* __restrict__ X, const bf16_t* __restrict__ W,
    const float* __restrict__ bias,
    bf16_t* __restrict__ Q, bf16_t* __restrict__ K, bf16_t* __restrict__ VT) {
  const int lane = threadIdx.x & 31;
  const int wid  = threadIdx.x >> 5;
  const int waveM = wid & 3;   // 4 waves along M
  const int waveN = wid >> 2;  // 2 waves along N
  const int rowBase = blockIdx.x * 128 + waveM * 32;
  const int colBase = blockIdx.y * 128 + waveN * 64;

  v8f acc[2][4];
#pragma unroll
  for (int mt = 0; mt < 2; ++mt)
#pragma unroll
    for (int nt = 0; nt < 4; ++nt) acc[mt][nt] = (v8f)0.0f;

  const bf16_t* Abase = X + (size_t)rowBase * C_;
  const bf16_t* Bbase = W + (size_t)colBase * C_;

  for (int k0 = 0; k0 < C_; k0 += 32) {
    v16bf a0 = load_a_frag(Abase + k0, C_, lane);
    v16bf a1 = load_a_frag(Abase + 16 * C_ + k0, C_, lane);
#pragma unroll
    for (int nt = 0; nt < 4; ++nt) {
      v16bf b = load_b_frag(Bbase + nt * 16 * C_ + k0, C_, lane);
      acc[0][nt] = wmma_bf16(a0, b, acc[0][nt]);
      acc[1][nt] = wmma_bf16(a1, b, acc[1][nt]);
    }
  }

  const int g = lane >> 4, nl = lane & 15;
#pragma unroll
  for (int mt = 0; mt < 2; ++mt) {
#pragma unroll
    for (int nt = 0; nt < 4; ++nt) {
      const int col  = colBase + nt * 16 + nl;  // 0..3071
      const float bv = bias[col];
      const int comp = col >> 10;               // 0=q,1=k,2=v
      const int hd   = col & 1023;
      const int h    = hd >> 6, d = hd & 63;
#pragma unroll
      for (int j = 0; j < 8; ++j) {
        const int m = rowBase + mt * 16 + g * 8 + j;   // 0..4095
        const int b = m >> 11, t = m & (T_ - 1);
        const int bh = b * NH_ + h;
        const bf16_t o = (bf16_t)(acc[mt][nt][j] + bv);
        if (comp == 0)      Q [((size_t)bh * T_ + t) * HD_ + d] = o;
        else if (comp == 1) K [((size_t)bh * T_ + t) * HD_ + d] = o;
        else                VT[((size_t)bh * HD_ + d) * T_ + t] = o;
      }
    }
  }
}

// ---------------------------------------------------------------------------
// Kernel 2: wave-level flash attention, online softmax in log2 domain.
// Each wave owns 32 query rows (two 16-row M tiles) so every K/V B-fragment
// feeds two WMMAs (halves the dominant L2 stream). P goes D-layout->A-layout
// through per-wave LDS tiles; next-block K/V prefetched to L0.
// ---------------------------------------------------------------------------
__global__ __launch_bounds__(256) void attn_kernel(
    const bf16_t* __restrict__ Q, const bf16_t* __restrict__ K,
    const bf16_t* __restrict__ VT, bf16_t* __restrict__ Y) {
  __shared__ bf16_t ldsP[8][2][16 * 32];   // per-wave two 16x32 bf16 P tiles

  const int lane = threadIdx.x & 31;
  const int wid  = threadIdx.x >> 5;
  const int bh   = blockIdx.x;                        // 0..31
  const int t0   = (blockIdx.y * 8 + wid) * 32;       // query tile start
  const int b    = bh >> 4, h = bh & 15;
  const int g    = lane >> 4, nl = lane & 15;

  const bf16_t* Qb = Q  + ((size_t)bh * T_ + t0) * HD_;
  const bf16_t* Kb = K  + (size_t)bh * T_ * HD_;
  const bf16_t* Vb = VT + (size_t)bh * HD_ * T_;

  // Loop-invariant Q fragments: [mt][d-half]
  v16bf aq[2][2];
#pragma unroll
  for (int mt = 0; mt < 2; ++mt) {
    aq[mt][0] = load_a_frag(Qb + mt * 16 * HD_,      HD_, lane);
    aq[mt][1] = load_a_frag(Qb + mt * 16 * HD_ + 32, HD_, lane);
  }

  v8f o[2][4];
#pragma unroll
  for (int mt = 0; mt < 2; ++mt)
#pragma unroll
    for (int dt = 0; dt < 4; ++dt) o[mt][dt] = (v8f)0.0f;

  float mrow[2][8], lrow[2][8];
#pragma unroll
  for (int mt = 0; mt < 2; ++mt)
#pragma unroll
    for (int j = 0; j < 8; ++j) {
      mrow[mt][j] = -__builtin_inff();
      lrow[mt][j] = 0.0f;
    }

  // logits pre-scaled into log2 domain: exp(s/8) == exp2(s * 0.125*log2(e))
  const float scale2 = 0.125f * 1.4426950408889634f;

  for (int s0 = 0; s0 <= t0 + 31; s0 += 32) {
    // ---- shared K B-fragments for this s-block ---------------------------
    v16bf bk00 = load_b_frag(Kb + (size_t)s0 * HD_,             HD_, lane);
    v16bf bk01 = load_b_frag(Kb + (size_t)s0 * HD_ + 32,        HD_, lane);
    v16bf bk10 = load_b_frag(Kb + (size_t)(s0 + 16) * HD_,      HD_, lane);
    v16bf bk11 = load_b_frag(Kb + (size_t)(s0 + 16) * HD_ + 32, HD_, lane);

    // ---- prefetch next block's K and V fragments into L0 -----------------
    if (s0 < t0) {
      const int s1 = s0 + 32;
      __builtin_prefetch(Kb + (size_t)(s1 + nl) * HD_ + g * 16, 0, 1);
      __builtin_prefetch(Kb + (size_t)(s1 + 16 + nl) * HD_ + g * 16, 0, 1);
#pragma unroll
      for (int dt = 0; dt < 4; ++dt)
        __builtin_prefetch(Vb + (size_t)(dt * 16 + nl) * T_ + s1 + g * 16,
                           0, 1);
    }

#pragma unroll
    for (int mt = 0; mt < 2; ++mt) {
      const int tmax = t0 + mt * 16 + 15;
      // ---- S = Q K^T for two 16-col tiles (skip fully-masked tiles) ------
      v8f sacc0 = (v8f)0.0f, sacc1 = (v8f)0.0f;
      if (s0 <= tmax) {
        sacc0 = wmma_bf16(aq[mt][0], bk00, sacc0);
        sacc0 = wmma_bf16(aq[mt][1], bk01, sacc0);
      }
      if (s0 + 16 <= tmax) {
        sacc1 = wmma_bf16(aq[mt][0], bk10, sacc1);
        sacc1 = wmma_bf16(aq[mt][1], bk11, sacc1);
      }

      // ---- online softmax (row spans 16 lanes; stats per VGPR row) -------
      bf16_t* myP = &ldsP[wid][mt][0];
#pragma unroll
      for (int j = 0; j < 8; ++j) {
        const int t  = t0 + mt * 16 + g * 8 + j;
        const int sA = s0 + nl, sB = s0 + 16 + nl;
        const float xA = (sA <= t) ? sacc0[j] * scale2 : -__builtin_inff();
        const float xB = (sB <= t) ? sacc1[j] * scale2 : -__builtin_inff();

        const float mnew = fmaxf(mrow[mt][j], rowmax16(fmaxf(xA, xB)));
        const float pA   = exp2f(xA - mnew);
        const float pB   = exp2f(xB - mnew);
        const float corr = exp2f(mrow[mt][j] - mnew);
        lrow[mt][j] = lrow[mt][j] * corr + rowsum16(pA + pB);
        mrow[mt][j] = mnew;
        o[mt][0][j] *= corr; o[mt][1][j] *= corr;
        o[mt][2][j] *= corr; o[mt][3][j] *= corr;

        // stage P (D layout -> LDS row-major 16x32)
        myP[(g * 8 + j) * 32 + nl]      = (bf16_t)pA;
        myP[(g * 8 + j) * 32 + 16 + nl] = (bf16_t)pB;
      }
    }
    asm volatile("s_wait_dscnt 0" ::: "memory");

    // ---- reload P tiles as A-fragments (same wave, in-order LDS) ---------
    v16bf ap[2];
#pragma unroll
    for (int mt = 0; mt < 2; ++mt) {
      const bf16_t* p = &ldsP[wid][mt][0] + (lane & 15) * 32 + g * 8;
      v8bf lo = *(const v8bf*)(p);
      v8bf hi = *(const v8bf*)(p + 16);
#pragma unroll
      for (int i = 0; i < 8; ++i) { ap[mt][i] = lo[i]; ap[mt][i + 8] = hi[i]; }
    }

    // ---- O += P V : 4 V B-frags shared across both M tiles ---------------
#pragma unroll
    for (int dt = 0; dt < 4; ++dt) {
      v16bf bv = load_b_frag(Vb + (size_t)(dt * 16) * T_ + s0, T_, lane);
      o[0][dt] = wmma_bf16(ap[0], bv, o[0][dt]);
      o[1][dt] = wmma_bf16(ap[1], bv, o[1][dt]);
    }
  }

  // ---- normalize and write y[b, t, h*64 + d] as bf16 ---------------------
#pragma unroll
  for (int mt = 0; mt < 2; ++mt) {
    float inv[8];
#pragma unroll
    for (int j = 0; j < 8; ++j) inv[j] = 1.0f / lrow[mt][j];
#pragma unroll
    for (int dt = 0; dt < 4; ++dt) {
#pragma unroll
      for (int j = 0; j < 8; ++j) {
        const int t = t0 + mt * 16 + g * 8 + j;
        Y[((size_t)b * T_ + t) * C_ + h * HD_ + dt * 16 + nl] =
            (bf16_t)(o[mt][dt][j] * inv[j]);
      }
    }
  }
}

// ---------------------------------------------------------------------------
// Kernel 3: output projection. [4096,1024] x [1024,1024]^T + bias -> f32 out.
// ---------------------------------------------------------------------------
__global__ __launch_bounds__(256) void proj_gemm_kernel(
    const bf16_t* __restrict__ Ym, const bf16_t* __restrict__ W,
    const float* __restrict__ bias, float* __restrict__ out) {
  const int lane = threadIdx.x & 31;
  const int wid  = threadIdx.x >> 5;
  const int waveM = wid & 3;
  const int waveN = wid >> 2;
  const int rowBase = blockIdx.x * 128 + waveM * 32;
  const int colBase = blockIdx.y * 128 + waveN * 64;

  v8f acc[2][4];
#pragma unroll
  for (int mt = 0; mt < 2; ++mt)
#pragma unroll
    for (int nt = 0; nt < 4; ++nt) acc[mt][nt] = (v8f)0.0f;

  const bf16_t* Abase = Ym + (size_t)rowBase * C_;
  const bf16_t* Bbase = W  + (size_t)colBase * C_;

  for (int k0 = 0; k0 < C_; k0 += 32) {
    v16bf a0 = load_a_frag(Abase + k0, C_, lane);
    v16bf a1 = load_a_frag(Abase + 16 * C_ + k0, C_, lane);
#pragma unroll
    for (int nt = 0; nt < 4; ++nt) {
      v16bf bfr = load_b_frag(Bbase + nt * 16 * C_ + k0, C_, lane);
      acc[0][nt] = wmma_bf16(a0, bfr, acc[0][nt]);
      acc[1][nt] = wmma_bf16(a1, bfr, acc[1][nt]);
    }
  }

  const int g = lane >> 4, nl = lane & 15;
#pragma unroll
  for (int mt = 0; mt < 2; ++mt) {
#pragma unroll
    for (int nt = 0; nt < 4; ++nt) {
      const int col = colBase + nt * 16 + nl;
      const float bv = bias[col];
#pragma unroll
      for (int j = 0; j < 8; ++j) {
        const int m = rowBase + mt * 16 + g * 8 + j;
        out[(size_t)m * C_ + col] = acc[mt][nt][j] + bv;
      }
    }
  }
}

// ---------------------------------------------------------------------------
// Host launcher
// ---------------------------------------------------------------------------
extern "C" void kernel_launch(void* const* d_in, const int* in_sizes, int n_in,
                              void* d_out, int out_size, void* d_ws,
                              size_t ws_size, hipStream_t stream) {
  (void)in_sizes; (void)n_in; (void)out_size; (void)ws_size;
  const float* x     = (const float*)d_in[0];
  const float* Wqkv  = (const float*)d_in[1];
  const float* bqkv  = (const float*)d_in[2];
  const float* Wproj = (const float*)d_in[3];
  const float* bproj = (const float*)d_in[4];
  float* out = (float*)d_out;

  char* p = (char*)d_ws;
  bf16_t* Xb  = (bf16_t*)p; p += (size_t)MTOK * C_ * 2;       // 8 MB
  bf16_t* Wqb = (bf16_t*)p; p += (size_t)3 * C_ * C_ * 2;     // 6 MB
  bf16_t* Wpb = (bf16_t*)p; p += (size_t)C_ * C_ * 2;         // 2 MB
  bf16_t* Qb  = (bf16_t*)p; p += (size_t)BH_ * T_ * HD_ * 2;  // 8 MB
  bf16_t* Kb  = (bf16_t*)p; p += (size_t)BH_ * T_ * HD_ * 2;  // 8 MB
  bf16_t* VTb = (bf16_t*)p; p += (size_t)BH_ * HD_ * T_ * 2;  // 8 MB
  bf16_t* Yb  = (bf16_t*)p; p += (size_t)MTOK * C_ * 2;       // 8 MB

  const int nX = MTOK * C_, nWq = 3 * C_ * C_, nWp = C_ * C_;
  cvt_f32_bf16_kernel<<<(nX  + 1023) / 1024, 256, 0, stream>>>(x, Xb, nX);
  cvt_f32_bf16_kernel<<<(nWq + 1023) / 1024, 256, 0, stream>>>(Wqkv, Wqb, nWq);
  cvt_f32_bf16_kernel<<<(nWp + 1023) / 1024, 256, 0, stream>>>(Wproj, Wpb, nWp);

  qkv_gemm_kernel<<<dim3(MTOK / 128, (3 * C_) / 128), 256, 0, stream>>>(
      Xb, Wqb, bqkv, Qb, Kb, VTb);

  attn_kernel<<<dim3(BH_, T_ / 256), 256, 0, stream>>>(Qb, Kb, VTb, Yb);

  proj_gemm_kernel<<<dim3(MTOK / 128, C_ / 128), 256, 0, stream>>>(
      Yb, Wpb, bproj, out);
}
